// RCNModel_31275951849780
// MI455X (gfx1250) — matrix operand order repaired and checked by reading, not compiled, not run on hardware
//
#include <hip/hip_runtime.h>

// ---------------------------------------------------------------------------
// GATv2 chess-GNN forward for MI455X (gfx1250, wave32, WMMA).
// Dense transforms use v_wmma_f32_16x16x32_f16 (f16 in, f32 acc).
// Edge softmax/aggregation in f32 with atomics; activations stored f16 to
// halve gather traffic (working set ~30MB -> resident in 192MB L2).
// ---------------------------------------------------------------------------

typedef __attribute__((ext_vector_type(16))) _Float16 v16h;
typedef __attribute__((ext_vector_type(8)))  _Float16 v8h;
typedef __attribute__((ext_vector_type(8)))  float    v8f;

#define NN    10000
#define NPAD  10016
#define EE    250000
#define GG    64

static __device__ __forceinline__ void atomicMaxF(float* addr, float v) {
  // IEEE-monotonic integer trick (works for mixed signs; init to -3e38)
  if (v >= 0.0f) atomicMax((int*)addr, __float_as_int(v));
  else           atomicMin((unsigned int*)addr, __float_as_uint(v));
}

// ---------------- utility fills -------------------------------------------
__global__ void k_fill(float* __restrict__ p, int n, float v) {
  int i = blockIdx.x * blockDim.x + threadIdx.x;
  if (i < n) p[i] = v;
}

// ---------------- weight packing into WMMA B-fragment order ----------------
// B (32x16 f16 per k-step), contiguous k order as in the A-frag: lane l holds N = l&15,
// K_local = 16*(l>>4) + i for vector element i in 0..15.
// Packed layout: Wp[(ct*ksteps + kt)*512 + lane*16 + i]
__global__ void k_pack_w(const float* __restrict__ W, _Float16* __restrict__ Wp,
                         int Kin, int Kpad) {
  int idx = blockIdx.x * blockDim.x + threadIdx.x;
  int total = Kpad * 256;
  if (idx >= total) return;
  int i    = idx & 15;
  int lane = (idx >> 4) & 31;
  int blk  = idx >> 9;
  int ksteps = Kpad >> 5;
  int ct = blk / ksteps;
  int kt = blk - ct * ksteps;
  int k = kt * 32 + ((lane >> 4) << 4) + i;
  int n = (ct << 4) + (lane & 15);
  float v = (k < Kin) ? W[k * 256 + n] : 0.0f;
  Wp[idx] = (_Float16)v;
}

// ---------------- edge-embedding table: ee[t][c] = edge_emb[t]@We + be -----
__global__ void k_ee_table(const float* __restrict__ edge_emb,
                           const float* __restrict__ We,
                           const float* __restrict__ be,
                           float* __restrict__ ee) {
  int idx = blockIdx.x * blockDim.x + threadIdx.x;   // 16*256
  if (idx >= 16 * 256) return;
  int t = idx >> 8, c = idx & 255;
  float s = be[c];
#pragma unroll 8
  for (int j = 0; j < 32; ++j) s += edge_emb[t * 32 + j] * We[j * 256 + c];
  ee[idx] = s;
}

// ---------------- input feature build: x_h[n][0:64] f16 (47 real, pad 0) ---
__global__ void k_build_x(const int* __restrict__ nt, const float* __restrict__ nf,
                          const float* __restrict__ node_emb,
                          _Float16* __restrict__ xh) {
  int idx = blockIdx.x * blockDim.x + threadIdx.x;
  if (idx >= NPAD * 64) return;
  int n = idx >> 6, k = idx & 63;
  float v = 0.0f;
  if (n < NN) {
    if (k < 32)      v = node_emb[nt[n] * 32 + k];
    else if (k < 47) v = nf[n * 15 + (k - 32)];
  }
  xh[idx] = (_Float16)v;
}

// ---------------- WMMA GEMM: Y[rows,256] = A[rows,Kpad] @ W + bias ---------
// One wave per 16x16 output tile. A-frag contiguous per lane half:
// lane<16: elems 0..15 -> K=kbase+0..15
// lane>=16: K=kbase+16..31.  B pre-packed contiguous per lane.
__global__ void k_gemm_wmma(const _Float16* __restrict__ A,
                            const _Float16* __restrict__ Wp,
                            const float* __restrict__ bias,
                            _Float16* __restrict__ Y, int Kpad) {
  const int lane = threadIdx.x & 31;
  const int rb = blockIdx.x << 4;
  const int ct = blockIdx.y;
  const int cb = ct << 4;
  const int ksteps = Kpad >> 5;
  const int m  = lane & 15;
  const int hi = lane >> 4;

  v8f acc = {};
  for (int kt = 0; kt < ksteps; ++kt) {
    const int kbase = kt << 5;
    const _Float16* ap = A + (size_t)(rb + m) * Kpad + kbase + hi * 16;
    v8h a0 = *(const v8h*)(ap);
    v8h a1 = *(const v8h*)(ap + 8);
    const _Float16* bp = Wp + (((size_t)(ct * ksteps + kt) << 5) + lane) * 16;
    v8h b0 = *(const v8h*)(bp);
    v8h b1 = *(const v8h*)(bp + 8);
    v16h av, bv;
#pragma unroll
    for (int i = 0; i < 8; ++i) {
      av[i] = a0[i]; av[8 + i] = a1[i];
      bv[i] = b0[i]; bv[8 + i] = b1[i];
    }
    acc = __builtin_amdgcn_wmma_f32_16x16x32_f16(false, av, false, bv,
                                                 (short)0, acc, false, false);
  }
  const int n = cb + m;
  const float bn_ = bias[n];
#pragma unroll
  for (int r = 0; r < 8; ++r) {
    const int row = rb + r + hi * 8;   // C layout: VGPR r, lanes<16 M=r, >=16 M=r+8
    Y[(size_t)row * 256 + n] = (_Float16)(acc[r] + bn_);
  }
}

// ---------------- edge pass 1: logits + per-(dst,head) max ----------------
// One wave per edge; lane owns channels [lane*8, lane*8+8), head = lane>>3.
__global__ void k_edge_logits(const int* __restrict__ ei, const int* __restrict__ ea,
                              const _Float16* __restrict__ xl, const _Float16* __restrict__ xr,
                              const float* __restrict__ ee, const float* __restrict__ att,
                              float* __restrict__ logits, float* __restrict__ nodemax) {
  int wid  = (blockIdx.x * blockDim.x + threadIdx.x) >> 5;
  int lane = threadIdx.x & 31;
  if (wid >= EE) return;
  int src = ei[wid], dst = ei[EE + wid], t = ea[wid];
  int ch = lane << 3;
  v8h xlv = *(const v8h*)(xl + (size_t)src * 256 + ch);
  v8h xrv = *(const v8h*)(xr + (size_t)dst * 256 + ch);
  const float* eep = ee + t * 256 + ch;
  const float* ap  = att + ch;        // att[H,C] flat == channel index
  float partial = 0.0f;
#pragma unroll
  for (int j = 0; j < 8; ++j) {
    float s = (float)xlv[j] + (float)xrv[j] + eep[j];
    s = (s > 0.0f) ? s : 0.2f * s;    // leaky_relu 0.2
    partial += s * ap[j];
  }
  partial += __shfl_xor(partial, 1);
  partial += __shfl_xor(partial, 2);
  partial += __shfl_xor(partial, 4);  // reduce within 8-lane head group
  if ((lane & 7) == 0) {
    int head = lane >> 3;
    logits[(size_t)wid * 4 + head] = partial;
    atomicMaxF(nodemax + dst * 4 + head, partial);
  }
}

// ---------------- edge pass 2: a = exp(l - max[dst]); den += a -------------
__global__ void k_edge_exp(const int* __restrict__ ei, const float* __restrict__ nodemax,
                           float* __restrict__ logits, float* __restrict__ den) {
  int idx = blockIdx.x * blockDim.x + threadIdx.x;
  if (idx >= EE * 4) return;
  int e = idx >> 2, h = idx & 3;
  int dst = ei[EE + e];
  float a = __expf(logits[idx] - nodemax[dst * 4 + h]);
  logits[idx] = a;
  atomicAdd(den + dst * 4 + h, a);
}

// ---------------- edge pass 3: agg[dst] += alpha * xl[src] ----------------
__global__ void k_edge_aggr(const int* __restrict__ ei, const _Float16* __restrict__ xl,
                            const float* __restrict__ logits, const float* __restrict__ den,
                            float* __restrict__ agg) {
  int wid  = (blockIdx.x * blockDim.x + threadIdx.x) >> 5;
  int lane = threadIdx.x & 31;
  if (wid >= EE) return;
  int src = ei[wid], dst = ei[EE + wid];
  int head = lane >> 3;
  float a = logits[(size_t)wid * 4 + head];
  float alpha = a / (den[dst * 4 + head] + 1e-16f);
  int ch = lane << 3;
  v8h xlv = *(const v8h*)(xl + (size_t)src * 256 + ch);
  float* op = agg + (size_t)dst * 256 + ch;
#pragma unroll
  for (int j = 0; j < 8; ++j) atomicAdd(op + j, alpha * (float)xlv[j]);
}

// ---------------- layer-1 node epilogue: +bias, BN, relu -> h1 f16 --------
__global__ void k_node1(const float* __restrict__ agg, const float* __restrict__ bias,
                        const float* __restrict__ gamma, const float* __restrict__ beta,
                        const float* __restrict__ mean, const float* __restrict__ var,
                        _Float16* __restrict__ h1) {
  int idx = blockIdx.x * blockDim.x + threadIdx.x;
  if (idx >= NPAD * 256) return;
  int n = idx >> 8, c = idx & 255;
  float v = 0.0f;
  if (n < NN) {
    float y = agg[idx] + bias[c];
    y = (y - mean[c]) * (gamma[c] * rsqrtf(var[c] + 1e-5f)) + beta[c];
    v = fmaxf(y, 0.0f);
  }
  h1[idx] = (_Float16)v;
}

// ---------------- layer-2 epilogue: head-mean, BN, relu, graph pooling ----
__global__ void k_node2_pool(const float* __restrict__ agg, const float* __restrict__ bias,
                             const float* __restrict__ gamma, const float* __restrict__ beta,
                             const float* __restrict__ mean, const float* __restrict__ var,
                             const int* __restrict__ batch,
                             float* __restrict__ gsum, float* __restrict__ counts) {
  int idx = blockIdx.x * blockDim.x + threadIdx.x;
  if (idx >= NN * 64) return;
  int n = idx >> 6, c = idx & 63;
  const float* an = agg + (size_t)n * 256;
  float y = 0.25f * (an[c] + an[64 + c] + an[128 + c] + an[192 + c]) + bias[c];
  y = (y - mean[c]) * (gamma[c] * rsqrtf(var[c] + 1e-5f)) + beta[c];
  y = fmaxf(y, 0.0f);
  int b = batch[n];
  atomicAdd(gsum + b * 64 + c, y);
  if (c == 0) atomicAdd(counts + b, 1.0f);
}

// ---------------- per-graph MLP heads -------------------------------------
__global__ void k_heads(const float* __restrict__ gsum, const float* __restrict__ counts,
                        const float* vW1, const float* vb1, const float* vW2, const float* vb2,
                        const float* pfW1, const float* pfb1, const float* pfW2, const float* pfb2,
                        const float* ptW1, const float* ptb1, const float* ptW2, const float* ptb2,
                        const float* tW1, const float* tb1, const float* tW2, const float* tb2,
                        const float* sW1, const float* sb1, const float* sW2, const float* sb2,
                        float* __restrict__ out) {
  __shared__ float gv[64];
  __shared__ float hid[5][128];
  int g = blockIdx.x, t = threadIdx.x;   // 128 threads
  float cnt = fmaxf(counts[g], 1.0f);
  if (t < 64) gv[t] = gsum[g * 64 + t] / cnt;
  __syncthreads();
  {
    float a0 = vb1[t], a1 = pfb1[t], a2 = ptb1[t], a3 = tb1[t], a4 = sb1[t];
    for (int j = 0; j < 64; ++j) {
      float x = gv[j];
      a0 += x * vW1[j * 128 + t];  a1 += x * pfW1[j * 128 + t];
      a2 += x * ptW1[j * 128 + t]; a3 += x * tW1[j * 128 + t];
      a4 += x * sW1[j * 128 + t];
    }
    hid[0][t] = fmaxf(a0, 0.f); hid[1][t] = fmaxf(a1, 0.f);
    hid[2][t] = fmaxf(a2, 0.f); hid[3][t] = fmaxf(a3, 0.f);
    hid[4][t] = fmaxf(a4, 0.f);
  }
  __syncthreads();
  if (t < 64) {
    float s = pfb2[t];
    for (int j = 0; j < 128; ++j) s += hid[1][j] * pfW2[j * 64 + t];
    out[64 + g * 64 + t] = s;                       // policy_from
  } else {
    int tt = t - 64;
    float s = ptb2[tt];
    for (int j = 0; j < 128; ++j) s += hid[2][j] * ptW2[j * 64 + tt];
    out[4160 + g * 64 + tt] = s;                    // policy_to
  }
  if (t == 0) {
    float s = vb2[0];
    for (int j = 0; j < 128; ++j) s += hid[0][j] * vW2[j];
    out[g] = tanhf(s);                              // value
  } else if (t == 1) {
    float s = tb2[0];
    for (int j = 0; j < 128; ++j) s += hid[3][j] * tW2[j];
    out[8256 + g] = 1.0f / (1.0f + __expf(-s));     // tactic
  } else if (t == 2) {
    float s = sb2[0];
    for (int j = 0; j < 128; ++j) s += hid[4][j] * sW2[j];
    out[8320 + g] = 1.0f / (1.0f + __expf(-s));     // strategic
  }
}

// ---------------------------------------------------------------------------
extern "C" void kernel_launch(void* const* d_in, const int* in_sizes, int n_in,
                              void* d_out, int out_size, void* d_ws, size_t ws_size,
                              hipStream_t stream) {
  (void)in_sizes; (void)n_in; (void)out_size; (void)ws_size;
  // ---- inputs (setup_inputs() dict order, nested dicts in insertion order)
  const int*   nt       = (const int*)d_in[0];
  const float* nf       = (const float*)d_in[1];
  const int*   ei       = (const int*)d_in[2];
  const int*   ea       = (const int*)d_in[3];
  const int*   batch    = (const int*)d_in[4];
  const float* node_emb = (const float*)d_in[5];
  const float* edge_emb = (const float*)d_in[6];
  const float* g1Wl = (const float*)d_in[7],  *g1bl = (const float*)d_in[8];
  const float* g1Wr = (const float*)d_in[9],  *g1br = (const float*)d_in[10];
  const float* g1We = (const float*)d_in[11], *g1be = (const float*)d_in[12];
  const float* g1att = (const float*)d_in[13], *g1bias = (const float*)d_in[14];
  const float* bn1g = (const float*)d_in[15], *bn1b = (const float*)d_in[16];
  const float* bn1m = (const float*)d_in[17], *bn1v = (const float*)d_in[18];
  const float* g2Wl = (const float*)d_in[19], *g2bl = (const float*)d_in[20];
  const float* g2Wr = (const float*)d_in[21], *g2br = (const float*)d_in[22];
  const float* g2We = (const float*)d_in[23], *g2be = (const float*)d_in[24];
  const float* g2att = (const float*)d_in[25], *g2bias = (const float*)d_in[26];
  const float* bn2g = (const float*)d_in[27], *bn2b = (const float*)d_in[28];
  const float* bn2m = (const float*)d_in[29], *bn2v = (const float*)d_in[30];
  const float* vW1 = (const float*)d_in[31], *vb1 = (const float*)d_in[32];
  const float* vW2 = (const float*)d_in[33], *vb2 = (const float*)d_in[34];
  const float* pfW1 = (const float*)d_in[35], *pfb1 = (const float*)d_in[36];
  const float* pfW2 = (const float*)d_in[37], *pfb2 = (const float*)d_in[38];
  const float* ptW1 = (const float*)d_in[39], *ptb1 = (const float*)d_in[40];
  const float* ptW2 = (const float*)d_in[41], *ptb2 = (const float*)d_in[42];
  const float* tW1 = (const float*)d_in[43], *tb1 = (const float*)d_in[44];
  const float* tW2 = (const float*)d_in[45], *tb2 = (const float*)d_in[46];
  const float* sW1 = (const float*)d_in[47], *sb1 = (const float*)d_in[48];
  const float* sW2 = (const float*)d_in[49], *sb2 = (const float*)d_in[50];
  float* out = (float*)d_out;

  // ---- workspace carve-up (256B aligned)
  char* ws = (char*)d_ws;
  size_t off = 0;
  auto take = [&](size_t bytes) -> char* {
    char* p = ws + off; off += (bytes + 255) & ~(size_t)255; return p;
  };
  _Float16* x_h  = (_Float16*)take((size_t)NPAD * 64 * 2);
  _Float16* h1_h = (_Float16*)take((size_t)NPAD * 256 * 2);
  _Float16* xl_h = (_Float16*)take((size_t)NPAD * 256 * 2);
  _Float16* xr_h = (_Float16*)take((size_t)NPAD * 256 * 2);
  _Float16* Wp1l = (_Float16*)take(64 * 256 * 2);
  _Float16* Wp1r = (_Float16*)take(64 * 256 * 2);
  _Float16* Wp2l = (_Float16*)take(256 * 256 * 2);
  _Float16* Wp2r = (_Float16*)take(256 * 256 * 2);
  float* ee1     = (float*)take(16 * 256 * 4);
  float* ee2     = (float*)take(16 * 256 * 4);
  float* logits  = (float*)take((size_t)EE * 4 * 4);
  float* nodemax = (float*)take((size_t)NN * 4 * 4);
  float* den     = (float*)take((size_t)NN * 4 * 4);
  float* agg     = (float*)take((size_t)NN * 256 * 4);
  float* gsum    = (float*)take(GG * 64 * 4);
  float* counts  = (float*)take(GG * 4);

  auto nb = [](long n) { return (int)((n + 255) / 256); };

  // ---- one-time prep (runs every launch; cheap, deterministic)
  k_pack_w<<<nb(64 * 256), 256, 0, stream>>>(g1Wl, Wp1l, 47, 64);
  k_pack_w<<<nb(64 * 256), 256, 0, stream>>>(g1Wr, Wp1r, 47, 64);
  k_pack_w<<<nb(256 * 256), 256, 0, stream>>>(g2Wl, Wp2l, 256, 256);
  k_pack_w<<<nb(256 * 256), 256, 0, stream>>>(g2Wr, Wp2r, 256, 256);
  k_ee_table<<<nb(16 * 256), 256, 0, stream>>>(edge_emb, g1We, g1be, ee1);
  k_ee_table<<<nb(16 * 256), 256, 0, stream>>>(edge_emb, g2We, g2be, ee2);
  k_build_x<<<nb((long)NPAD * 64), 256, 0, stream>>>(nt, nf, node_emb, x_h);

  // ---- init accumulators (layer 1)
  k_fill<<<nb((long)NN * 256), 256, 0, stream>>>(agg, NN * 256, 0.0f);
  k_fill<<<nb(NN * 4), 256, 0, stream>>>(den, NN * 4, 0.0f);
  k_fill<<<nb(NN * 4), 256, 0, stream>>>(nodemax, NN * 4, -3.0e38f);
  k_fill<<<nb(GG * 64), 256, 0, stream>>>(gsum, GG * 64, 0.0f);
  k_fill<<<nb(GG), 256, 0, stream>>>(counts, GG, 0.0f);

  // ---- layer 1
  k_gemm_wmma<<<dim3(NPAD / 16, 16), 32, 0, stream>>>(x_h, Wp1l, g1bl, xl_h, 64);
  k_gemm_wmma<<<dim3(NPAD / 16, 16), 32, 0, stream>>>(x_h, Wp1r, g1br, xr_h, 64);
  k_edge_logits<<<nb((long)EE * 32), 256, 0, stream>>>(ei, ea, xl_h, xr_h, ee1, g1att,
                                                       logits, nodemax);
  k_edge_exp<<<nb((long)EE * 4), 256, 0, stream>>>(ei, nodemax, logits, den);
  k_edge_aggr<<<nb((long)EE * 32), 256, 0, stream>>>(ei, xl_h, logits, den, agg);
  k_node1<<<nb((long)NPAD * 256), 256, 0, stream>>>(agg, g1bias, bn1g, bn1b, bn1m, bn1v, h1_h);

  // ---- re-init accumulators (layer 2)
  k_fill<<<nb((long)NN * 256), 256, 0, stream>>>(agg, NN * 256, 0.0f);
  k_fill<<<nb(NN * 4), 256, 0, stream>>>(den, NN * 4, 0.0f);
  k_fill<<<nb(NN * 4), 256, 0, stream>>>(nodemax, NN * 4, -3.0e38f);

  // ---- layer 2
  k_gemm_wmma<<<dim3(NPAD / 16, 16), 32, 0, stream>>>(h1_h, Wp2l, g2bl, xl_h, 256);
  k_gemm_wmma<<<dim3(NPAD / 16, 16), 32, 0, stream>>>(h1_h, Wp2r, g2br, xr_h, 256);
  k_edge_logits<<<nb((long)EE * 32), 256, 0, stream>>>(ei, ea, xl_h, xr_h, ee2, g2att,
                                                       logits, nodemax);
  k_edge_exp<<<nb((long)EE * 4), 256, 0, stream>>>(ei, nodemax, logits, den);
  k_edge_aggr<<<nb((long)EE * 32), 256, 0, stream>>>(ei, xl_h, logits, den, agg);
  k_node2_pool<<<nb((long)NN * 64), 256, 0, stream>>>(agg, g2bias, bn2g, bn2b, bn2m, bn2v,
                                                      batch, gsum, counts);

  // ---- heads
  k_heads<<<GG, 128, 0, stream>>>(gsum, counts,
                                  vW1, vb1, vW2, vb2,
                                  pfW1, pfb1, pfW2, pfb2,
                                  ptW1, ptb1, ptW2, ptb2,
                                  tW1, tb1, tW2, tb2,
                                  sW1, sb1, sW2, sb2,
                                  out);
}
